// MVSNet_17282948399140
// MI455X (gfx1250) — compile-verified, matrix operand order checked by reading, not run on hardware
//
#include <hip/hip_runtime.h>

typedef __attribute__((ext_vector_type(16))) _Float16 v16h;
typedef __attribute__((ext_vector_type(8)))  _Float16 v8h;
typedef __attribute__((ext_vector_type(8)))  float    v8f;

#define MVS_STEPS 256

// ============================================================
// Cost volume: for each (depth step, pixel) warp src via homography,
// bilinear sample, L1 vs ref, write f16 channel into NHWC x0 concat buf.
// ============================================================
__global__ void cost_volume_kernel(const float* __restrict__ ref,
                                   const float* __restrict__ src,
                                   const float* __restrict__ KRKiUV,
                                   const float* __restrict__ Kt,
                                   _Float16* __restrict__ x0,
                                   int H, int W, int Cp)
{
    int N = H * W;
    int idx = blockIdx.x * blockDim.x + threadIdx.x;
    if (idx >= MVS_STEPS * N) return;
    int step = idx / N;
    int p = idx - step * N;
    int y = p / W;
    int x = p - y * W;

    const float idmin = 1.0f / 50.0f, idmax = 2.0f;
    const float idstep = (idmax - idmin) / (float)(MVS_STEPS - 1);
    float depth = 1.0f / (idmin + (float)step * idstep);

    int q = x * H + y;  // KRKiUV was built with meshgrid(W,H,'ij')
    float wu = depth * KRKiUV[q]         + Kt[0];
    float wv = depth * KRKiUV[N + q]     + Kt[1];
    float wz = depth * KRKiUV[2 * N + q] + Kt[2];
    float inv = 1.0f / (wz + 1e-6f);
    float sx = wu * inv - 0.5f;   // (uv-nb)/nb then grid->pix collapses
    float sy = wv * inv - 0.5f;

    float x0f = floorf(sx), y0f = floorf(sy);
    float wx = sx - x0f, wy = sy - y0f;
    float w00 = (1.f - wx) * (1.f - wy);
    float w01 = wx * (1.f - wy);
    float w10 = (1.f - wx) * wy;
    float w11 = wx * wy;

    float cost = 0.f;
#pragma unroll
    for (int c = 0; c < 3; ++c) {
        const float* sc = src + c * N;
        auto tap = [&](float yf, float xf) -> float {
            bool valid = (xf >= 0.f) && (xf <= (float)(W - 1)) &&
                         (yf >= 0.f) && (yf <= (float)(H - 1));
            int xi = (int)fminf(fmaxf(xf, 0.f), (float)(W - 1));
            int yi = (int)fminf(fmaxf(yf, 0.f), (float)(H - 1));
            return valid ? sc[yi * W + xi] : 0.f;
        };
        float warped = tap(y0f, x0f)             * w00 +
                       tap(y0f, x0f + 1.f)       * w01 +
                       tap(y0f + 1.f, x0f)       * w10 +
                       tap(y0f + 1.f, x0f + 1.f) * w11;
        cost += fabsf(ref[c * N + p] - warped);
    }
    x0[(size_t)p * Cp + 3 + step] = (_Float16)cost;
}

// ============================================================
// Drop the 3 reference channels into NHWC x0.
// ============================================================
__global__ void ref_to_x0_kernel(const float* __restrict__ ref,
                                 _Float16* __restrict__ x0, int N, int Cp)
{
    int idx = blockIdx.x * blockDim.x + threadIdx.x;
    if (idx >= 3 * N) return;
    int c = idx / N;
    int p = idx - c * N;
    x0[(size_t)p * Cp + c] = (_Float16)ref[idx];
}

__global__ void zero_f16_kernel(_Float16* __restrict__ p, int n)
{
    int i = blockIdx.x * blockDim.x + threadIdx.x;
    if (i < n) p[i] = (_Float16)0.f;
}

// ============================================================
// Weight repack: [Cout][Cin][ks][ks] f32 -> [tap][Cout][Cp] f16, zero pad.
// ============================================================
__global__ void wrepack_kernel(const float* __restrict__ w,
                               _Float16* __restrict__ wp,
                               int Cout, int Cin, int ks, int Cp)
{
    int kk = ks * ks;
    int total = kk * Cout * Cp;
    int idx = blockIdx.x * blockDim.x + threadIdx.x;
    if (idx >= total) return;
    int ci  = idx % Cp;
    int r   = idx / Cp;
    int mo  = r % Cout;
    int tap = r / Cout;
    int ky = tap / ks, kx = tap - ky * ks;
    float v = (ci < Cin) ? w[((mo * Cin + ci) * ks + ky) * ks + kx] : 0.f;
    wp[idx] = (_Float16)v;
}

// ============================================================
// Implicit-GEMM conv, WMMA f32 += f16 x f16.
// One wave computes a 32(out-ch) x 64(pixel) tile:
//   2 M-subtiles x 4 N-subtiles = 8 f32 accumulators.
// Per K-step (K=32 input channels, per (ky,kx) tap):
//   4 A-fragment b128 loads + 8 B-fragment b128 loads -> 8 v_wmma
//   (B fragments reused across both M rows back-to-back).
// ============================================================
__global__ __launch_bounds__(32) void conv_wmma_kernel(
    const _Float16* __restrict__ act,   // [Hin*Win][Cp] NHWC f16
    const _Float16* __restrict__ wp,    // [ks*ks][Cout][Cp] f16
    float*          __restrict__ out,   // [Cout][Hout*Wout] f32 raw
    int Cp, int Cout, int Hin, int Win, int Hout, int Wout,
    int ks, int stride)
{
    const int lane = threadIdx.x;
    const int lm   = lane & 15;
    const int half = lane >> 4;
    const int Np   = Hout * Wout;
    const int mbase = blockIdx.y * 32;
    const int nbase = blockIdx.x * 64;
    const int pad  = (ks - 1) >> 1;
    const int tapStride = Cout * Cp;

    const int m0 = mbase + lm;
    const int m1 = mbase + 16 + lm;
    const bool mv0 = m0 < Cout;
    const bool mv1 = m1 < Cout;

    int oy[4], ox[4];
    bool nvalid[4];
#pragma unroll
    for (int t = 0; t < 4; ++t) {
        int n = nbase + 16 * t + lm;
        nvalid[t] = n < Np;
        int nn = nvalid[t] ? n : 0;
        oy[t] = nn / Wout;
        ox[t] = nn - oy[t] * Wout;
    }

    v8f acc0[4] = {};
    v8f acc1[4] = {};
    const _Float16* wbase0 = wp + (size_t)(mv0 ? m0 : 0) * Cp;
    const _Float16* wbase1 = wp + (size_t)(mv1 ? m1 : 0) * Cp;

    for (int ky = 0; ky < ks; ++ky) {
        for (int kx = 0; kx < ks; ++kx) {
            const _Float16* wrow0 = wbase0;
            const _Float16* wrow1 = wbase1;
            wbase0 += tapStride;
            wbase1 += tapStride;

            const _Float16* arow[4];
            bool pvalid[4];
#pragma unroll
            for (int t = 0; t < 4; ++t) {
                int iy = oy[t] * stride + ky - pad;
                int ix = ox[t] * stride + kx - pad;
                pvalid[t] = nvalid[t] && iy >= 0 && iy < Hin &&
                            ix >= 0 && ix < Win;
                arow[t] = act + (size_t)(pvalid[t] ? (iy * Win + ix) : 0) * Cp;
            }

            for (int ci = 0; ci < Cp; ci += 32) {
                v8h zero = {};
                v8h a0lo = mv0 ? *(const v8h*)(wrow0 + ci + half * 8)      : zero;
                v8h a0hi = mv0 ? *(const v8h*)(wrow0 + ci + 16 + half * 8) : zero;
                v8h a1lo = mv1 ? *(const v8h*)(wrow1 + ci + half * 8)      : zero;
                v8h a1hi = mv1 ? *(const v8h*)(wrow1 + ci + 16 + half * 8) : zero;
                v16h a0 = __builtin_shufflevector(a0lo, a0hi,
                    0,1,2,3,4,5,6,7,8,9,10,11,12,13,14,15);
                v16h a1 = __builtin_shufflevector(a1lo, a1hi,
                    0,1,2,3,4,5,6,7,8,9,10,11,12,13,14,15);
#pragma unroll
                for (int t = 0; t < 4; ++t) {
                    v8h blo = pvalid[t] ? *(const v8h*)(arow[t] + ci + half * 8)      : zero;
                    v8h bhi = pvalid[t] ? *(const v8h*)(arow[t] + ci + 16 + half * 8) : zero;
                    v16h b = __builtin_shufflevector(blo, bhi,
                        0,1,2,3,4,5,6,7,8,9,10,11,12,13,14,15);
                    acc0[t] = __builtin_amdgcn_wmma_f32_16x16x32_f16(
                        false, a0, false, b, (short)0, acc0[t], false, false);
                    acc1[t] = __builtin_amdgcn_wmma_f32_16x16x32_f16(
                        false, a1, false, b, (short)0, acc1[t], false, false);
                }
            }
        }
    }

#pragma unroll
    for (int t = 0; t < 4; ++t) {
        int n = nbase + 16 * t + lm;
        if (n < Np) {
#pragma unroll
            for (int v = 0; v < 8; ++v) {
                int mmA = mbase + half * 8 + v;
                int mmB = mbase + 16 + half * 8 + v;
                if (mmA < Cout) out[(size_t)mmA * Np + n] = acc0[t][v];
                if (mmB < Cout) out[(size_t)mmB * Np + n] = acc1[t][v];
            }
        }
    }
}

// ============================================================
// BatchNorm statistics: one block per channel, LDS tree reduction.
// (raw conv output is [Cout][Np] row-major -> contiguous rows)
// ============================================================
__global__ void bn_stats_kernel(const float* __restrict__ x, int N,
                                float* __restrict__ mean,
                                float* __restrict__ var)
{
    int c = blockIdx.x;
    const float* xc = x + (size_t)c * N;
    float s = 0.f, s2 = 0.f;
    for (int i = threadIdx.x; i < N; i += 256) {
        float v = xc[i];
        s += v; s2 += v * v;
    }
    __shared__ float sh[256];
    __shared__ float sh2[256];
    sh[threadIdx.x] = s; sh2[threadIdx.x] = s2;
    __syncthreads();
    for (int off = 128; off > 0; off >>= 1) {
        if ((int)threadIdx.x < off) {
            sh[threadIdx.x]  += sh[threadIdx.x + off];
            sh2[threadIdx.x] += sh2[threadIdx.x + off];
        }
        __syncthreads();
    }
    if (threadIdx.x == 0) {
        float mv = sh[0] / (float)N;
        mean[c] = mv;
        var[c]  = sh2[0] / (float)N - mv * mv;
    }
}

// ============================================================
// BN + ReLU apply: f32 raw [C][N] -> f16 NHWC dst (stride Cd, offset choff).
// ============================================================
__global__ void bn_relu_kernel(const float* __restrict__ x,
                               const float* __restrict__ mean,
                               const float* __restrict__ var,
                               const float* __restrict__ g,
                               const float* __restrict__ b,
                               _Float16* __restrict__ dst,
                               int C, int N, int Cd, int choff)
{
    int idx = blockIdx.x * blockDim.x + threadIdx.x;
    if (idx >= C * N) return;
    int c = idx / N;
    int p = idx - c * N;
    float y = (x[idx] - mean[c]) * rsqrtf(var[c] + 1e-5f) * g[c] + b[c];
    dst[(size_t)p * Cd + choff + c] = (_Float16)fmaxf(y, 0.f);
}

// ============================================================
// Bilinear x2 upsample (linspace style), NHWC f16 -> NHWC f16.
// ============================================================
__global__ void up2_kernel(const _Float16* __restrict__ src,
                           _Float16* __restrict__ dst,
                           int C, int h, int w, int Cs, int Cd, int choff)
{
    int ho = 2 * h, wo = 2 * w;
    int No = ho * wo;
    int idx = blockIdx.x * blockDim.x + threadIdx.x;
    if (idx >= C * No) return;
    int c = idx / No;
    int p = idx - c * No;
    int oy = p / wo;
    int ox = p - oy * wo;

    float ys = (float)oy * (float)(h - 1) / (float)(ho - 1);
    float xs = (float)ox * (float)(w - 1) / (float)(wo - 1);
    int y0 = (int)floorf(ys); y0 = y0 < 0 ? 0 : (y0 > h - 2 ? h - 2 : y0);
    int x0 = (int)floorf(xs); x0 = x0 < 0 ? 0 : (x0 > w - 2 ? w - 2 : x0);
    float wy = ys - (float)y0;
    float wx = xs - (float)x0;

    const _Float16* sc = src + c;
    float v00 = (float)sc[(size_t)(y0 * w + x0) * Cs];
    float v01 = (float)sc[(size_t)(y0 * w + x0 + 1) * Cs];
    float v10 = (float)sc[(size_t)((y0 + 1) * w + x0) * Cs];
    float v11 = (float)sc[(size_t)((y0 + 1) * w + x0 + 1) * Cs];
    float val = (v00 * (1.f - wy) + v10 * wy) * (1.f - wx) +
                (v01 * (1.f - wy) + v11 * wy) * wx;
    dst[(size_t)p * Cd + choff + c] = (_Float16)val;
}

// ============================================================
// NHWC channel-block copy into concat buffer.
// ============================================================
__global__ void copy_ch_kernel(const _Float16* __restrict__ src,
                               _Float16* __restrict__ dst,
                               int C, int N, int Cs, int Cd, int choff)
{
    int idx = blockIdx.x * blockDim.x + threadIdx.x;
    if (idx >= C * N) return;
    int p = idx / C;
    int c = idx - p * C;
    dst[(size_t)p * Cd + choff + c] = src[(size_t)p * Cs + c];
}

// ============================================================
// Disparity head: 2*sigmoid(conv + bias); f32 to d_out, f16 for later up2.
// ============================================================
__global__ void disp_kernel(const float* __restrict__ raw,
                            const float* __restrict__ bias,
                            float* __restrict__ out_f32,
                            _Float16* __restrict__ out_f16, int N)
{
    int i = blockIdx.x * blockDim.x + threadIdx.x;
    if (i >= N) return;
    float v = 2.0f / (1.0f + expf(-(raw[i] + bias[0])));
    out_f32[i] = v;
    out_f16[i] = (_Float16)v;
}

// ============================================================
// Host orchestration
// ============================================================
extern "C" void kernel_launch(void* const* d_in, const int* in_sizes, int n_in,
                              void* d_out, int out_size, void* d_ws, size_t ws_size,
                              hipStream_t stream)
{
    (void)in_sizes; (void)n_in; (void)out_size; (void)ws_size;

    const int H0 = 128, W0 = 160;
    const int H1 = 64,  W1 = 80;
    const int H2 = 32,  W2 = 40;
    const int H3 = 16,  W3 = 20;
    const int H4 = 8,   W4 = 10;
    const int H5 = 4,   W5 = 5;
    const int N0 = H0 * W0, N1 = H1 * W1, N2 = H2 * W2,
              N3 = H3 * W3, N4 = H4 * W4, N5 = H5 * W5;

    // ---- inputs ----
    const float* ref    = (const float*)d_in[0];
    const float* srcimg = (const float*)d_in[1];
    const float* KRKiUV = (const float*)d_in[2];
    const float* Kt     = (const float*)d_in[3];
    int pi = 4;
    auto PP = [&]() { return (const float*)d_in[pi++]; };
    const float *c1w1 = PP(), *c1g1 = PP(), *c1b1 = PP(), *c1w2 = PP(), *c1g2 = PP(), *c1b2 = PP();
    const float *c2w1 = PP(), *c2g1 = PP(), *c2b1 = PP(), *c2w2 = PP(), *c2g2 = PP(), *c2b2 = PP();
    const float *c3w1 = PP(), *c3g1 = PP(), *c3b1 = PP(), *c3w2 = PP(), *c3g2 = PP(), *c3b2 = PP();
    const float *c4w1 = PP(), *c4g1 = PP(), *c4b1 = PP(), *c4w2 = PP(), *c4g2 = PP(), *c4b2 = PP();
    const float *c5w1 = PP(), *c5g1 = PP(), *c5b1 = PP(), *c5w2 = PP(), *c5g2 = PP(), *c5b2 = PP();
    const float *u4w = PP(), *u4g = PP(), *u4b = PP();
    const float *u3w = PP(), *u3g = PP(), *u3b = PP();
    const float *u2w = PP(), *u2g = PP(), *u2b = PP();
    const float *u1w = PP(), *u1g = PP(), *u1b = PP();
    const float *u0w = PP(), *u0g = PP(), *u0b = PP();
    const float *i4w = PP(), *i4g = PP(), *i4b = PP();
    const float *i3w = PP(), *i3g = PP(), *i3b = PP();
    const float *i2w = PP(), *i2g = PP(), *i2b = PP();
    const float *i1w = PP(), *i1g = PP(), *i1b = PP();
    const float *i0w = PP(), *i0g = PP(), *i0b = PP();
    const float *d3w = PP(), *d3b = PP();
    const float *d2w = PP(), *d2b = PP();
    const float *d1w = PP(), *d1b = PP();
    const float *d0w = PP(), *d0b = PP();

    // ---- output slices (return order d0,d1,d2,d3) ----
    float* out_d0 = (float*)d_out;
    float* out_d1 = out_d0 + N0;
    float* out_d2 = out_d1 + N1;
    float* out_d3 = out_d2 + N2;

    // ---- workspace bump allocator ----
    char* ws = (char*)d_ws;
    size_t off = 0;
    auto alloc = [&](size_t bytes) -> void* {
        void* p = ws + off;
        off += (bytes + 255) & ~(size_t)255;
        return p;
    };
    float*    raw   = (float*)alloc(sizeof(float) * 128 * N0);
    float*    meanb = (float*)alloc(sizeof(float) * 1024);
    float*    varb  = (float*)alloc(sizeof(float) * 1024);
    _Float16* wpack = (_Float16*)alloc(sizeof(_Float16) * 5 * 1024 * 1024);
    _Float16* x0    = (_Float16*)alloc(sizeof(_Float16) * (size_t)N0 * 288);
    _Float16* c1a   = (_Float16*)alloc(sizeof(_Float16) * (size_t)N0 * 128);
    _Float16* c1    = (_Float16*)alloc(sizeof(_Float16) * (size_t)N1 * 128);
    _Float16* c2a   = (_Float16*)alloc(sizeof(_Float16) * (size_t)N1 * 256);
    _Float16* c2    = (_Float16*)alloc(sizeof(_Float16) * (size_t)N2 * 256);
    _Float16* c3a   = (_Float16*)alloc(sizeof(_Float16) * (size_t)N2 * 512);
    _Float16* c3    = (_Float16*)alloc(sizeof(_Float16) * (size_t)N3 * 512);
    _Float16* c4a   = (_Float16*)alloc(sizeof(_Float16) * (size_t)N3 * 512);
    _Float16* c4    = (_Float16*)alloc(sizeof(_Float16) * (size_t)N4 * 512);
    _Float16* c5a   = (_Float16*)alloc(sizeof(_Float16) * (size_t)N4 * 512);
    _Float16* c5    = (_Float16*)alloc(sizeof(_Float16) * (size_t)N5 * 512);
    _Float16* up2t  = (_Float16*)alloc(sizeof(_Float16) * (size_t)N0 * 128);
    _Float16* cat4  = (_Float16*)alloc(sizeof(_Float16) * (size_t)N4 * 1024);
    _Float16* i4    = (_Float16*)alloc(sizeof(_Float16) * (size_t)N4 * 512);
    _Float16* cat3  = (_Float16*)alloc(sizeof(_Float16) * (size_t)N3 * 1024);
    _Float16* i3    = (_Float16*)alloc(sizeof(_Float16) * (size_t)N3 * 512);
    _Float16* cat2  = (_Float16*)alloc(sizeof(_Float16) * (size_t)N2 * 544);
    _Float16* i2    = (_Float16*)alloc(sizeof(_Float16) * (size_t)N2 * 256);
    _Float16* cat1  = (_Float16*)alloc(sizeof(_Float16) * (size_t)N1 * 288);
    _Float16* i1    = (_Float16*)alloc(sizeof(_Float16) * (size_t)N1 * 128);
    _Float16* cat0  = (_Float16*)alloc(sizeof(_Float16) * (size_t)N0 * 96);
    _Float16* i0    = (_Float16*)alloc(sizeof(_Float16) * (size_t)N0 * 64);
    _Float16* d3h   = (_Float16*)alloc(sizeof(_Float16) * N3);
    _Float16* d2h   = (_Float16*)alloc(sizeof(_Float16) * N2);
    _Float16* d1h   = (_Float16*)alloc(sizeof(_Float16) * N1);
    _Float16* d0h   = (_Float16*)alloc(sizeof(_Float16) * N0);

    // ---- helpers ----
    auto zero = [&](_Float16* p, int n) {
        zero_f16_kernel<<<(n + 255) / 256, 256, 0, stream>>>(p, n);
    };
    auto conv = [&](const _Float16* a, const float* w, float* o,
                    int Cin, int Cp, int Cout, int Hin, int Win,
                    int Hout, int Wout, int ks, int st) {
        int wtot = ks * ks * Cout * Cp;
        wrepack_kernel<<<(wtot + 255) / 256, 256, 0, stream>>>(
            w, wpack, Cout, Cin, ks, Cp);
        dim3 g((Hout * Wout + 63) / 64, (Cout + 31) / 32, 1);
        conv_wmma_kernel<<<g, dim3(32), 0, stream>>>(
            a, wpack, o, Cp, Cout, Hin, Win, Hout, Wout, ks, st);
    };
    auto bnrelu = [&](const float* rawp, const float* g, const float* b,
                      _Float16* dst, int C, int N, int Cd, int ch) {
        bn_stats_kernel<<<C, 256, 0, stream>>>(rawp, N, meanb, varb);
        int tot = C * N;
        bn_relu_kernel<<<(tot + 255) / 256, 256, 0, stream>>>(
            rawp, meanb, varb, g, b, dst, C, N, Cd, ch);
    };
    auto up2 = [&](const _Float16* s, _Float16* dst, int C, int h, int w,
                   int Cs, int Cd, int ch) {
        int tot = C * 4 * h * w;
        up2_kernel<<<(tot + 255) / 256, 256, 0, stream>>>(s, dst, C, h, w, Cs, Cd, ch);
    };
    auto copych = [&](const _Float16* s, _Float16* dst, int C, int N,
                      int Cs, int Cd, int ch) {
        int tot = C * N;
        copy_ch_kernel<<<(tot + 255) / 256, 256, 0, stream>>>(s, dst, C, N, Cs, Cd, ch);
    };
    auto disp = [&](const _Float16* feat, int Cs, const float* w, const float* b,
                    float* of32, _Float16* of16, int h, int wd) {
        int N = h * wd;
        conv(feat, w, raw, Cs, Cs, 1, h, wd, h, wd, 3, 1);
        disp_kernel<<<(N + 255) / 256, 256, 0, stream>>>(raw, b, of32, of16, N);
    };

    // ---- zero padded concat buffers (pad channels must stay finite) ----
    zero(x0,   N0 * 288);
    zero(cat2, N2 * 544);
    zero(cat1, N1 * 288);
    zero(cat0, N0 * 96);

    // ---- cost volume + input concat (NHWC, Cp=288) ----
    ref_to_x0_kernel<<<(3 * N0 + 255) / 256, 256, 0, stream>>>(ref, x0, N0, 288);
    cost_volume_kernel<<<(MVS_STEPS * N0 + 255) / 256, 256, 0, stream>>>(
        ref, srcimg, KRKiUV, Kt, x0, H0, W0, 288);

    // ---- encoder ----
    conv(x0,  c1w1, raw, 259, 288, 128, H0, W0, H0, W0, 7, 1); bnrelu(raw, c1g1, c1b1, c1a, 128, N0, 128, 0);
    conv(c1a, c1w2, raw, 128, 128, 128, H0, W0, H1, W1, 5, 2); bnrelu(raw, c1g2, c1b2, c1,  128, N1, 128, 0);
    conv(c1,  c2w1, raw, 128, 128, 256, H1, W1, H1, W1, 5, 1); bnrelu(raw, c2g1, c2b1, c2a, 256, N1, 256, 0);
    conv(c2a, c2w2, raw, 256, 256, 256, H1, W1, H2, W2, 5, 2); bnrelu(raw, c2g2, c2b2, c2,  256, N2, 256, 0);
    conv(c2,  c3w1, raw, 256, 256, 512, H2, W2, H2, W2, 3, 1); bnrelu(raw, c3g1, c3b1, c3a, 512, N2, 512, 0);
    conv(c3a, c3w2, raw, 512, 512, 512, H2, W2, H3, W3, 3, 2); bnrelu(raw, c3g2, c3b2, c3,  512, N3, 512, 0);
    conv(c3,  c4w1, raw, 512, 512, 512, H3, W3, H3, W3, 3, 1); bnrelu(raw, c4g1, c4b1, c4a, 512, N3, 512, 0);
    conv(c4a, c4w2, raw, 512, 512, 512, H3, W3, H4, W4, 3, 2); bnrelu(raw, c4g2, c4b2, c4,  512, N4, 512, 0);
    conv(c4,  c5w1, raw, 512, 512, 512, H4, W4, H4, W4, 3, 1); bnrelu(raw, c5g1, c5b1, c5a, 512, N4, 512, 0);
    conv(c5a, c5w2, raw, 512, 512, 512, H4, W4, H5, W5, 3, 2); bnrelu(raw, c5g2, c5b2, c5,  512, N5, 512, 0);

    // ---- decoder level 4 ----
    up2(c5, up2t, 512, H5, W5, 512, 512, 0);
    conv(up2t, u4w, raw, 512, 512, 512, H4, W4, H4, W4, 3, 1);
    bnrelu(raw, u4g, u4b, cat4, 512, N4, 1024, 0);
    copych(c4, cat4, 512, N4, 512, 1024, 512);
    conv(cat4, i4w, raw, 1024, 1024, 512, H4, W4, H4, W4, 3, 1);
    bnrelu(raw, i4g, i4b, i4, 512, N4, 512, 0);

    // ---- level 3 ----
    up2(i4, up2t, 512, H4, W4, 512, 512, 0);
    conv(up2t, u3w, raw, 512, 512, 512, H3, W3, H3, W3, 3, 1);
    bnrelu(raw, u3g, u3b, cat3, 512, N3, 1024, 0);
    copych(c3, cat3, 512, N3, 512, 1024, 512);
    conv(cat3, i3w, raw, 1024, 1024, 512, H3, W3, H3, W3, 3, 1);
    bnrelu(raw, i3g, i3b, i3, 512, N3, 512, 0);
    disp(i3, 512, d3w, d3b, out_d3, d3h, H3, W3);

    // ---- level 2 (concat: up(i3) 256 | c2 256 | up2(d3) 1 = 513 -> 544) ----
    up2(i3, up2t, 512, H3, W3, 512, 512, 0);
    conv(up2t, u2w, raw, 512, 512, 256, H2, W2, H2, W2, 3, 1);
    bnrelu(raw, u2g, u2b, cat2, 256, N2, 544, 0);
    copych(c2, cat2, 256, N2, 256, 544, 256);
    up2(d3h, cat2, 1, H3, W3, 1, 544, 512);
    conv(cat2, i2w, raw, 513, 544, 256, H2, W2, H2, W2, 3, 1);
    bnrelu(raw, i2g, i2b, i2, 256, N2, 256, 0);
    disp(i2, 256, d2w, d2b, out_d2, d2h, H2, W2);

    // ---- level 1 (concat: up(i2) 128 | c1 128 | up2(d2) 1 = 257 -> 288) ----
    up2(i2, up2t, 256, H2, W2, 256, 256, 0);
    conv(up2t, u1w, raw, 256, 256, 128, H1, W1, H1, W1, 3, 1);
    bnrelu(raw, u1g, u1b, cat1, 128, N1, 288, 0);
    copych(c1, cat1, 128, N1, 128, 288, 128);
    up2(d2h, cat1, 1, H2, W2, 1, 288, 256);
    conv(cat1, i1w, raw, 257, 288, 128, H1, W1, H1, W1, 3, 1);
    bnrelu(raw, i1g, i1b, i1, 128, N1, 128, 0);
    disp(i1, 128, d1w, d1b, out_d1, d1h, H1, W1);

    // ---- level 0 (concat: up(i1) 64 | up2(d1) 1 = 65 -> 96) ----
    up2(i1, up2t, 128, H1, W1, 128, 128, 0);
    conv(up2t, u0w, raw, 128, 128, 64, H0, W0, H0, W0, 3, 1);
    bnrelu(raw, u0g, u0b, cat0, 64, N0, 96, 0);
    up2(d1h, cat0, 1, H1, W1, 1, 96, 64);
    conv(cat0, i0w, raw, 65, 96, 64, H0, W0, H0, W0, 3, 1);
    bnrelu(raw, i0g, i0b, i0, 64, N0, 64, 0);
    disp(i0, 64, d0w, d0b, out_d0, d0h, H0, W0);
}
